// Mamba_45535243272446
// MI455X (gfx1250) — matrix operand order repaired and checked by reading, not compiled, not run on hardware
//
#include <hip/hip_runtime.h>
#include <hip/hip_bf16.h>

#define D_MODEL 1024
#define D_STATE 16
#define D_CONV  4
#define D_INNER 2048
#define DT_RANK 64
#define BATCH   2
#define SEQ     2048
#define MROWS   (BATCH * SEQ)   /* 4096 flattened (b,l) rows */
#define NCHUNK  16
#define CHLEN   (SEQ / NCHUNK)  /* 128 steps per scan chunk */

typedef __bf16 bf16_t;
typedef __bf16 v16bf __attribute__((ext_vector_type(16)));
typedef __bf16 v8bf  __attribute__((ext_vector_type(8)));
typedef float  v8f   __attribute__((ext_vector_type(8)));

// ---------- helpers ----------
static __device__ __forceinline__ bf16_t f2bf(float f) {
  unsigned u; __builtin_memcpy(&u, &f, 4);
  unsigned r = u + 0x7FFFu + ((u >> 16) & 1u);   // round-to-nearest-even
  unsigned short hs = (unsigned short)(r >> 16);
  bf16_t b; __builtin_memcpy(&b, &hs, 2); return b;
}
static __device__ __forceinline__ float bf2f(bf16_t b) {
  unsigned short s; __builtin_memcpy(&s, &b, 2);
  unsigned u = ((unsigned)s) << 16;
  float f; __builtin_memcpy(&f, &u, 4); return f;
}
static __device__ __forceinline__ float silu_f(float x) { return x / (1.f + __expf(-x)); }
static __device__ __forceinline__ float softplus_f(float x) {
  return (x > 20.f) ? x : log1pf(__expf(x));
}
static __device__ __forceinline__ v16bf load_afrag(const bf16_t* p) {
  v8bf lo = *(const v8bf*)p;
  v8bf hi = *(const v8bf*)(p + 8);
  return __builtin_shufflevector(lo, hi,
      0, 1, 2, 3, 4, 5, 6, 7, 8, 9, 10, 11, 12, 13, 14, 15);
}

// ---------- fp32 -> bf16 packing ----------
__global__ __launch_bounds__(256) void f32_to_bf16_kernel(
    const float* __restrict__ s, bf16_t* __restrict__ d, long n) {
  long i = (long)blockIdx.x * 256 + threadIdx.x;
  if (i < n) d[i] = f2bf(s[i]);
}
__global__ __launch_bounds__(256) void f32_to_bf16_pad_kernel(
    const float* __restrict__ s, bf16_t* __restrict__ d, long nsrc, long ntot) {
  long i = (long)blockIdx.x * 256 + threadIdx.x;
  if (i < ntot) d[i] = (i < nsrc) ? f2bf(s[i]) : f2bf(0.f);
}

// ---------- bf16 WMMA GEMM: C(M,N) = A(M,K) @ W(N,K)^T, fp32 accum ----------
// One wave -> 32(M) x 64(N) tile (2 A-frags x 4 B-frags = 8 WMMAs / k-step).
// Block = 8 waves stacked in M -> 256(M) x 64(N). grid.x = ceil(N/64), grid.y = M/256.
// MODE 0: in_proj  epilogue -> x_raw fp32 (n<2048) | z_silu bf16 (n>=2048)
// MODE 1: x_proj   epilogue -> x_dbl fp32 (n<96)   & dtlo bf16 (n<64)
// MODE 2: dt_proj  epilogue -> delta = softplus(acc + 2*bias[n]) fp32
// MODE 3: out_proj epilogue -> fp32 output
template<int MODE>
__global__ __launch_bounds__(256) void gemm_bf16_wmma(
    const bf16_t* __restrict__ A, const bf16_t* __restrict__ W, int K,
    float* __restrict__ out0, bf16_t* __restrict__ out1,
    const float* __restrict__ bias) {
  const int lane = threadIdx.x & 31;
  const int wave = threadIdx.x >> 5;
  const int m0 = blockIdx.y * 256 + wave * 32;
  const int n0 = blockIdx.x * 64;

  v8f zero = {0.f, 0.f, 0.f, 0.f, 0.f, 0.f, 0.f, 0.f};
  v8f acc[2][4] = {{zero, zero, zero, zero}, {zero, zero, zero, zero}};

  // A-fragment: lane -> row m+(lane&15); 16 contiguous ks at (lane<16?0:16)
  const long arow0 = (long)(m0 + (lane & 15)) * K + ((lane & 16) ? 16 : 0);
  const long arow1 = arow0 + (long)16 * K;
  // B-fragment: lane -> weight row n+(lane&15); 16 contiguous ks at (lane<16?0:16)
  const long brow0 = (long)(n0 + (lane & 15)) * K + ((lane & 16) ? 16 : 0);

  for (int k0 = 0; k0 < K; k0 += 32) {
    v16bf av0 = load_afrag(A + arow0 + k0);
    v16bf av1 = load_afrag(A + arow1 + k0);
    __builtin_prefetch(A + arow0 + k0 + 32, 0, 1);
    __builtin_prefetch(A + arow1 + k0 + 32, 0, 1);
#pragma unroll
    for (int t = 0; t < 4; ++t) {
      v16bf bv = *(const v16bf*)(W + brow0 + (long)(16 * t) * K + k0);
      acc[0][t] = __builtin_amdgcn_wmma_f32_16x16x32_bf16(
          false, av0, false, bv, (short)0, acc[0][t], false, false);
      acc[1][t] = __builtin_amdgcn_wmma_f32_16x16x32_bf16(
          false, av1, false, bv, (short)0, acc[1][t], false, false);
    }
  }

  // C/D layout: lane -> n = lane&15 ; VGPR r -> m = r + (lane<16?0:8)
  const int cn  = lane & 15;
  const int cmb = (lane >> 4) * 8;
#pragma unroll
  for (int mi = 0; mi < 2; ++mi) {
#pragma unroll
    for (int t = 0; t < 4; ++t) {
      const int n = n0 + 16 * t + cn;
#pragma unroll
      for (int r = 0; r < 8; ++r) {
        const int m = m0 + mi * 16 + cmb + r;
        const float v = acc[mi][t][r];
        if (MODE == 0) {
          if (n < D_INNER) out0[(long)m * D_INNER + n] = v;
          else             out1[(long)m * D_INNER + (n - D_INNER)] = f2bf(silu_f(v));
        } else if (MODE == 1) {
          if (n < DT_RANK + 2 * D_STATE) out0[(long)m * 96 + n] = v;
          if (n < DT_RANK)               out1[(long)m * DT_RANK + n] = f2bf(v);
        } else if (MODE == 2) {
          out0[(long)m * D_INNER + n] = softplus_f(v + 2.f * bias[n]);
        } else {
          out0[(long)m * D_MODEL + n] = v;
        }
      }
    }
  }
}

// ---------- causal depthwise conv1d (k=4) + bias + SiLU ----------
__global__ __launch_bounds__(256) void conv_silu_kernel(
    const float* __restrict__ x, const float* __restrict__ cw,
    const float* __restrict__ cb,
    float* __restrict__ y32, bf16_t* __restrict__ y16) {
  long idx = (long)blockIdx.x * 256 + threadIdx.x;   // over MROWS*D_INNER
  int d = (int)(idx & (D_INNER - 1));
  long row = idx >> 11;                               // b*SEQ + l
  int l = (int)(row & (SEQ - 1));
  const float* w = cw + d * D_CONV;
  float acc = cb[d];
#pragma unroll
  for (int t = 0; t < D_CONV; ++t) {
    int ls = l - (D_CONV - 1) + t;
    if (ls >= 0) acc += w[t] * x[(row - (D_CONV - 1) + t) * (long)D_INNER + d];
  }
  float s = silu_f(acc);
  y32[idx] = s;
  y16[idx] = f2bf(s);
}

// ---------- chunked selective scan (associative recurrence h <- a*h + b) ----------
// Phase 1: per (chunk, channel) run 128 steps from h=0; emit per-state decay
//          product aprod[16] and final state hfin[16]. 16x seq-parallel.
// Phase 2: tiny sequential chain over 16 chunks -> true initial state hinit.
// Phase 3: rerun each chunk from hinit, emit y fused with D*u skip + silu(z) gate.
// grid for P1/P3: (NCHUNK, 16 channel-groups) x 256 threads; ch = b*2048 + d.
__global__ __launch_bounds__(256) void scan_part1(
    const float* __restrict__ delta, const float* __restrict__ u,
    const float* __restrict__ xdbl, const float* __restrict__ A_log,
    float* __restrict__ aprod, float* __restrict__ hfin) {
  const int c = blockIdx.x;
  const int g = blockIdx.y;
  const int b = g >> 3;
  const int d = (g & 7) * 256 + threadIdx.x;
  const int ch = b * D_INNER + d;

  float A[D_STATE], h[D_STATE], p[D_STATE];
#pragma unroll
  for (int n = 0; n < D_STATE; ++n) {
    A[n] = -__expf(A_log[(long)d * D_STATE + n]);
    h[n] = 0.f; p[n] = 1.f;
  }

  __shared__ float sB[16][D_STATE];
  const long rowbase = (long)b * SEQ + (long)c * CHLEN;
  for (int l0 = 0; l0 < CHLEN; l0 += 16) {
    __syncthreads();
    {
      int s = threadIdx.x >> 4, cc = threadIdx.x & 15;     // 256 floats, 1/thread
      sB[s][cc] = xdbl[(rowbase + l0 + s) * 96 + DT_RANK + cc];
    }
    __syncthreads();
#pragma unroll 1
    for (int s = 0; s < 16; ++s) {
      const long off = (rowbase + l0 + s) * (long)D_INNER + d;
      const float dl = delta[off];
      const float du = dl * u[off];
#pragma unroll
      for (int n = 0; n < D_STATE; ++n) {
        float a = __expf(dl * A[n]);
        p[n] *= a;
        h[n] = a * h[n] + du * sB[s][n];
      }
    }
  }
  const long base = ((long)c * MROWS + ch) * D_STATE;
#pragma unroll
  for (int n = 0; n < D_STATE; ++n) { aprod[base + n] = p[n]; hfin[base + n] = h[n]; }
}

__global__ __launch_bounds__(256) void scan_part2(
    const float* __restrict__ aprod, const float* __restrict__ hfin,
    float* __restrict__ hinit) {
  const int ch = blockIdx.x * 256 + threadIdx.x;   // 0..4095
  float h[D_STATE];
#pragma unroll
  for (int n = 0; n < D_STATE; ++n) h[n] = 0.f;
#pragma unroll 1
  for (int c = 0; c < NCHUNK; ++c) {
    const long base = ((long)c * MROWS + ch) * D_STATE;
#pragma unroll
    for (int n = 0; n < D_STATE; ++n) {
      hinit[base + n] = h[n];
      h[n] = aprod[base + n] * h[n] + hfin[base + n];
    }
  }
}

__global__ __launch_bounds__(256) void scan_part3(
    const float* __restrict__ delta, const float* __restrict__ u,
    const float* __restrict__ xdbl, const bf16_t* __restrict__ zsilu,
    const float* __restrict__ A_log, const float* __restrict__ Dp,
    const float* __restrict__ hinit, bf16_t* __restrict__ ybf) {
  const int c = blockIdx.x;
  const int g = blockIdx.y;
  const int b = g >> 3;
  const int d = (g & 7) * 256 + threadIdx.x;
  const int ch = b * D_INNER + d;

  float A[D_STATE], h[D_STATE];
  const long base = ((long)c * MROWS + ch) * D_STATE;
#pragma unroll
  for (int n = 0; n < D_STATE; ++n) {
    A[n] = -__expf(A_log[(long)d * D_STATE + n]);
    h[n] = hinit[base + n];
  }
  const float Dd = Dp[d];

  __shared__ float sB[16][D_STATE];
  __shared__ float sC[16][D_STATE];
  const long rowbase = (long)b * SEQ + (long)c * CHLEN;
  for (int l0 = 0; l0 < CHLEN; l0 += 16) {
    __syncthreads();
    for (int i = threadIdx.x; i < 16 * 32; i += 256) {     // 512 floats, 2/thread
      int s = i >> 5, cc = i & 31;
      float v = xdbl[(rowbase + l0 + s) * 96 + DT_RANK + cc];
      if (cc < D_STATE) sB[s][cc] = v; else sC[s][cc - D_STATE] = v;
    }
    __syncthreads();
#pragma unroll 1
    for (int s = 0; s < 16; ++s) {
      const long off = (rowbase + l0 + s) * (long)D_INNER + d;
      const float dl = delta[off];
      const float ut = u[off];
      const float du = dl * ut;
      float y = Dd * ut;
#pragma unroll
      for (int n = 0; n < D_STATE; ++n) {
        h[n] = __expf(dl * A[n]) * h[n] + du * sB[s][n];
        y += h[n] * sC[s][n];
      }
      ybf[off] = f2bf(y * bf2f(zsilu[off]));
    }
  }
}

// ---------- launch ----------
extern "C" void kernel_launch(void* const* d_in, const int* in_sizes, int n_in,
                              void* d_out, int out_size, void* d_ws, size_t ws_size,
                              hipStream_t stream) {
  const float* hidden = (const float*)d_in[0];
  const float* w_in   = (const float*)d_in[1];
  const float* conv_w = (const float*)d_in[2];
  const float* conv_b = (const float*)d_in[3];
  const float* w_xp   = (const float*)d_in[4];
  const float* w_dt   = (const float*)d_in[5];
  const float* dt_b   = (const float*)d_in[6];
  const float* A_log  = (const float*)d_in[7];
  const float* Dp     = (const float*)d_in[8];
  const float* w_out  = (const float*)d_in[9];
  float* outp = (float*)d_out;

  char* p = (char*)d_ws;
  auto alloc = [&](size_t bytes) -> char* {
    char* r = p; p += (bytes + 255) & ~(size_t)255; return r;
  };
  bf16_t* hid_bf  = (bf16_t*)alloc((size_t)MROWS * D_MODEL * 2);
  bf16_t* win_bf  = (bf16_t*)alloc((size_t)(2 * D_INNER) * D_MODEL * 2);
  bf16_t* wxp_bf  = (bf16_t*)alloc((size_t)128 * D_INNER * 2);        // padded 96->128 rows
  bf16_t* wdt_bf  = (bf16_t*)alloc((size_t)D_INNER * DT_RANK * 2);
  bf16_t* wout_bf = (bf16_t*)alloc((size_t)D_MODEL * D_INNER * 2);
  float*  x_raw   = (float*)alloc((size_t)MROWS * D_INNER * 4);
  bf16_t* z_silu  = (bf16_t*)alloc((size_t)MROWS * D_INNER * 2);
  float*  xconv   = (float*)alloc((size_t)MROWS * D_INNER * 4);
  bf16_t* xconvbf = (bf16_t*)alloc((size_t)MROWS * D_INNER * 2);
  float*  x_dbl   = (float*)alloc((size_t)MROWS * 96 * 4);
  bf16_t* dtlo_bf = (bf16_t*)alloc((size_t)MROWS * DT_RANK * 2);
  float*  deltab  = (float*)alloc((size_t)MROWS * D_INNER * 4);
  bf16_t* y_bf    = (bf16_t*)alloc((size_t)MROWS * D_INNER * 2);
  float*  aprod   = (float*)alloc((size_t)NCHUNK * MROWS * D_STATE * 4);
  float*  hfin    = (float*)alloc((size_t)NCHUNK * MROWS * D_STATE * 4);
  float*  hinit   = (float*)alloc((size_t)NCHUNK * MROWS * D_STATE * 4);

  dim3 blk(256);
  auto cvt = [&](const float* s, bf16_t* dst, long n) {
    f32_to_bf16_kernel<<<dim3((unsigned)((n + 255) / 256)), blk, 0, stream>>>(s, dst, n);
  };
  cvt(hidden, hid_bf, (long)MROWS * D_MODEL);
  cvt(w_in,   win_bf, (long)(2 * D_INNER) * D_MODEL);
  f32_to_bf16_pad_kernel<<<dim3((unsigned)((128L * D_INNER + 255) / 256)), blk, 0, stream>>>(
      w_xp, wxp_bf, 96L * D_INNER, 128L * D_INNER);
  cvt(w_dt,  wdt_bf,  (long)D_INNER * DT_RANK);
  cvt(w_out, wout_bf, (long)D_MODEL * D_INNER);

  // xz = hidden @ in_proj^T : split into x (fp32) and silu(z) (bf16)
  gemm_bf16_wmma<0><<<dim3(2 * D_INNER / 64, MROWS / 256), blk, 0, stream>>>(
      hid_bf, win_bf, D_MODEL, x_raw, z_silu, nullptr);

  // causal depthwise conv + bias + SiLU
  conv_silu_kernel<<<dim3((unsigned)((long)MROWS * D_INNER / 256)), blk, 0, stream>>>(
      x_raw, conv_w, conv_b, xconv, xconvbf);

  // x_dbl = x @ x_proj^T (N=96, grid covers padded 128)
  gemm_bf16_wmma<1><<<dim3(2, MROWS / 256), blk, 0, stream>>>(
      xconvbf, wxp_bf, D_INNER, x_dbl, dtlo_bf, nullptr);

  // delta = softplus(dt_lo @ dt_proj^T + 2*dt_proj_b)
  gemm_bf16_wmma<2><<<dim3(D_INNER / 64, MROWS / 256), blk, 0, stream>>>(
      dtlo_bf, wdt_bf, DT_RANK, deltab, nullptr, dt_b);

  // chunked selective scan
  scan_part1<<<dim3(NCHUNK, 16), blk, 0, stream>>>(
      deltab, xconv, x_dbl, A_log, aprod, hfin);
  scan_part2<<<dim3(MROWS / 256), blk, 0, stream>>>(aprod, hfin, hinit);
  scan_part3<<<dim3(NCHUNK, 16), blk, 0, stream>>>(
      deltab, xconv, x_dbl, z_silu, A_log, Dp, hinit, y_bf);

  // out = y @ out_proj^T (fp32)
  gemm_bf16_wmma<3><<<dim3(D_MODEL / 64, MROWS / 256), blk, 0, stream>>>(
      y_bf, wout_bf, D_INNER, outp, nullptr, nullptr);
}
